// GraphConvolutionLayer_11716670784206
// MI455X (gfx1250) — matrix operand order repaired and checked by reading, not compiled, not run on hardware
//
#include <hip/hip_runtime.h>
#include <hip/hip_bf16.h>

typedef __attribute__((ext_vector_type(16))) __bf16        v16bf;
typedef __attribute__((ext_vector_type(8)))  float         v8f;
typedef __attribute__((ext_vector_type(8)))  unsigned int  v8u;

#define D_IN  256
#define D_OUT 128
#define A_STRIDE 264   // ushorts; 528 B = 132 dwords, %64 banks = 4 -> conflict-free b128 column reads

// ---- f32 -> (bf16 hi, bf16 lo) split, round-to-nearest-even -----------------
__device__ __forceinline__ unsigned int bf16_hi_bits(float f) {
    unsigned int u = __float_as_uint(f);
    return (u + 0x7FFFu + ((u >> 16) & 1u)) >> 16;
}
__device__ __forceinline__ float bf16_bits_to_f32(unsigned int b) {
    return __uint_as_float(b << 16);
}

union Frag { v8u u; v16bf b; };

// ---------------------------------------------------------------------------
// Kernel 0: split + transpose weight once: w[k][n] f32 -> w_hi/w_lo[n][k] bf16
// 128 KB total, L2-resident, reused by every GEMM block.
// ---------------------------------------------------------------------------
__global__ __launch_bounds__(256) void split_weight(
    const float* __restrict__ w,
    unsigned short* __restrict__ w_hi, unsigned short* __restrict__ w_lo)
{
    int i = blockIdx.x * 256 + threadIdx.x;      // 0 .. 32767
    int k = i >> 7;                              // row in [0,256)
    int n = i & (D_OUT - 1);                     // col in [0,128)
    float f = w[i];
    unsigned int hb = bf16_hi_bits(f);
    float        lo = f - bf16_bits_to_f32(hb);
    w_hi[n * D_IN + k] = (unsigned short)hb;
    w_lo[n * D_IN + k] = (unsigned short)bf16_hi_bits(lo);
}

// ---------------------------------------------------------------------------
// Kernel 1: support = (x .* mask) @ W  via bf16x3 on V_WMMA_F32_16X16X32_BF16
// Block: 256 thr = 8 waves; block = 16 rows, wave w = 16-col tile of D_OUT.
// K=256 swept in 8 chunks of 32; 3 WMMAs per chunk (hi*hi + hi*lo + lo*hi).
// ---------------------------------------------------------------------------
__global__ __launch_bounds__(256) void gemm_wmma_bf16x3(
    const float* __restrict__ x, const float* __restrict__ mask,
    const unsigned short* __restrict__ w_hi, const unsigned short* __restrict__ w_lo,
    float* __restrict__ support)
{
    __shared__ unsigned short a_hi[16 * A_STRIDE];
    __shared__ unsigned short a_lo[16 * A_STRIDE];

    const int tid  = threadIdx.x;
    const int row0 = blockIdx.x * 16;

    // Stage dropout-applied A tile (16x256), split into bf16 hi/lo. Coalesced.
    #pragma unroll
    for (int i = 0; i < 16; ++i) {
        int idx = tid + i * 256;
        int r = idx >> 8;
        int k = idx & (D_IN - 1);
        int g = (row0 + r) * D_IN + k;
        float f  = x[g] * mask[g];
        unsigned int hb = bf16_hi_bits(f);
        float        lo = f - bf16_bits_to_f32(hb);
        a_hi[r * A_STRIDE + k] = (unsigned short)hb;
        a_lo[r * A_STRIDE + k] = (unsigned short)bf16_hi_bits(lo);
    }
    __syncthreads();

    const int wave = tid >> 5;          // 0..7 : column tile
    const int lane = tid & 31;
    const int r    = lane & 15;         // A row / B col within tile
    const int h    = lane >> 4;         // lane half
    const int gcol = wave * 16 + r;     // global output column

    v8f acc = {};
    #pragma unroll
    for (int t = 0; t < D_IN / 32; ++t) {
        const int kb = t * 32;

        // A 16x32 bf16 layout: V0-3 <-> K=2v+8h, V4-7 <-> K=16+2(v-4)+8h
        // => two contiguous 16 B runs at ushort offsets kb+8h and kb+16+8h.
        const uint4* pah = (const uint4*)&a_hi[r * A_STRIDE + kb + 8 * h];
        const uint4* pal = (const uint4*)&a_lo[r * A_STRIDE + kb + 8 * h];
        uint4 ah0 = pah[0], ah1 = pah[2];    // +2 uint4 = +16 ushorts
        uint4 al0 = pal[0], al1 = pal[2];

        // B 32x16 bf16 layout: same K order as A (V0-3 <-> K=2v+8h, V4-7 <-> K=16+2(v-4)+8h)
        // => two contiguous 16 B runs in transposed weight at [gcol][kb+8h] and [gcol][kb+16+8h].
        const uint4* pbh = (const uint4*)&w_hi[gcol * D_IN + kb + 8 * h];
        const uint4* pbl = (const uint4*)&w_lo[gcol * D_IN + kb + 8 * h];
        uint4 bh0 = pbh[0], bh1 = pbh[2];
        uint4 bl0 = pbl[0], bl1 = pbl[2];

        Frag fah, fal, fbh, fbl;
        fah.u = (v8u){ah0.x, ah0.y, ah0.z, ah0.w, ah1.x, ah1.y, ah1.z, ah1.w};
        fal.u = (v8u){al0.x, al0.y, al0.z, al0.w, al1.x, al1.y, al1.z, al1.w};
        fbh.u = (v8u){bh0.x, bh0.y, bh0.z, bh0.w, bh1.x, bh1.y, bh1.z, bh1.w};
        fbl.u = (v8u){bl0.x, bl0.y, bl0.z, bl0.w, bl1.x, bl1.y, bl1.z, bl1.w};

        // bf16x3: A*B ~= Ahi*Bhi + Ahi*Blo + Alo*Bhi   (f32 accumulate)
        acc = __builtin_amdgcn_wmma_f32_16x16x32_bf16(
                  false, fah.b, false, fbh.b, (short)0, acc, false, false);
        acc = __builtin_amdgcn_wmma_f32_16x16x32_bf16(
                  false, fah.b, false, fbl.b, (short)0, acc, false, false);
        acc = __builtin_amdgcn_wmma_f32_16x16x32_bf16(
                  false, fal.b, false, fbh.b, (short)0, acc, false, false);
    }

    // C/D layout: VGPR i -> M=i (lanes 0-15) / M=i+8 (lanes 16-31); N = lane&15
    const int rbase = row0 + (h << 3);
    #pragma unroll
    for (int i = 0; i < 8; ++i)
        support[(rbase + i) * D_OUT + gcol] = acc[i];
}

// ---------------------------------------------------------------------------
// Kernel 2: zero the output accumulator (harness poisons d_out with 0xAA)
// ---------------------------------------------------------------------------
__global__ __launch_bounds__(256) void zero_out(float4* __restrict__ out, int n4)
{
    int i = blockIdx.x * blockDim.x + threadIdx.x;
    if (i < n4) out[i] = make_float4(0.f, 0.f, 0.f, 0.f);
}

// ---------------------------------------------------------------------------
// Kernel 3: COO scatter  out[row] += val * support[col]
// One wave per edge: 32 lanes x float4 = 128 features. support & out are both
// L2-resident (51.2 MB each < 192 MB L2); fp32 atomics resolve at L2
// (global_atomic_add_f32, no-return form on STOREcnt).
// ---------------------------------------------------------------------------
__global__ __launch_bounds__(256) void spmm_scatter(
    const float* __restrict__ support, const float* __restrict__ eval,
    const int* __restrict__ erow, const int* __restrict__ ecol,
    float* __restrict__ out, int E)
{
    int e = blockIdx.x * 8 + threadIdx.y;
    if (e >= E) return;

    const int   r = erow[e];
    const int   c = ecol[e];
    const float v = eval[e];

    const float4* src = (const float4*)(support + (size_t)c * D_OUT);
    float*        dst = out + (size_t)r * D_OUT;

    const int d = threadIdx.x;          // 0..31 -> features [4d, 4d+4)
    float4 s = src[d];
    unsafeAtomicAdd(dst + 4 * d + 0, v * s.x);
    unsafeAtomicAdd(dst + 4 * d + 1, v * s.y);
    unsafeAtomicAdd(dst + 4 * d + 2, v * s.z);
    unsafeAtomicAdd(dst + 4 * d + 3, v * s.w);
}

// ---------------------------------------------------------------------------
extern "C" void kernel_launch(void* const* d_in, const int* in_sizes, int n_in,
                              void* d_out, int out_size, void* d_ws, size_t ws_size,
                              hipStream_t stream)
{
    const float* x    = (const float*)d_in[0];
    const float* mask = (const float*)d_in[1];
    const float* w    = (const float*)d_in[2];
    const float* ev   = (const float*)d_in[3];
    const int*   er   = (const int*)d_in[4];
    const int*   ec   = (const int*)d_in[5];
    float*       out  = (float*)d_out;

    // workspace layout: [w_hi 64KB][w_lo 64KB][support 51.2MB]
    unsigned short* w_hi = (unsigned short*)d_ws;
    unsigned short* w_lo = w_hi + D_IN * D_OUT;
    float*          supp = (float*)((char*)d_ws + 2u * D_IN * D_OUT * sizeof(unsigned short));

    const int N = in_sizes[0] / D_IN;      // 100000 (multiple of 16)
    const int E = in_sizes[3];             // 1600000

    // 0) one-shot weight split+transpose to bf16 hi/lo
    split_weight<<<(D_IN * D_OUT) / 256, 256, 0, stream>>>(w, w_hi, w_lo);

    // 1) fused dropout + bf16x3 GEMM -> support in workspace
    gemm_wmma_bf16x3<<<N / 16, 256, 0, stream>>>(x, mask, w_hi, w_lo, supp);

    // 2) clear output
    const int n4 = out_size / 4;
    zero_out<<<(n4 + 255) / 256, 256, 0, stream>>>((float4*)out, n4);

    // 3) edge scatter with hardware fp32 atomics
    spmm_scatter<<<(E + 7) / 8, dim3(32, 8), 0, stream>>>(supp, ev, er, ec, out, E);
}
